// coAttention_alt_10582799417983
// MI455X (gfx1250) — compile-verified, matrix operand order checked by reading, not run on hardware
//
#include <hip/hip_runtime.h>
#include <math.h>

#define B_    4096
#define R_    49
#define S_    50
#define D_    512
#define K_    100
#define MPAD  64     // rows padded to 4 tiles of 16
#define NTILE 7      // 112 cols = 7 tiles of 16 (K=100 padded)
#define MTILE 4
#define KB_   (D_ / 32)          // 16 k-blocks
#define ATTN_THREADS 224         // 7 waves: one N-tile per wave
#define NPREP (NTILE * KB_ * 32 * 8)   // dwords in a prepped weight matrix (28672)

typedef __attribute__((ext_vector_type(16))) __bf16 v16bf;
typedef __attribute__((ext_vector_type(8)))  float  v8f;
typedef __attribute__((ext_vector_type(4)))  unsigned int v4u;
typedef __attribute__((ext_vector_type(8)))  unsigned int v8u;
typedef __attribute__((ext_vector_type(8)))  int v8i;
typedef __attribute__((ext_vector_type(4)))  int v4i;

#if defined(__gfx1250__) && __has_builtin(__builtin_amdgcn_tensor_load_to_lds) && \
    __has_builtin(__builtin_amdgcn_s_wait_tensorcnt)
#define USE_TDM 1
#else
#define USE_TDM 0
#endif

#if defined(__gfx1250__) && __has_builtin(__builtin_amdgcn_sched_group_barrier)
// Pattern per k-step: 2 VMEM reads (B frag), 8 DS reads (A frags), 4 WMMAs.
#define SCHED_HINT()                                        \
  do {                                                      \
    __builtin_amdgcn_sched_group_barrier(0x020, 2, 0);      \
    __builtin_amdgcn_sched_group_barrier(0x100, 8, 0);      \
    __builtin_amdgcn_sched_group_barrier(0x008, 4, 0);      \
  } while (0)
#else
#define SCHED_HINT() do {} while (0)
#endif

__device__ __forceinline__ unsigned int pack_bf16(float f0, float f1) {
  __bf16 b0 = (__bf16)f0, b1 = (__bf16)f1;
  return (unsigned int)__builtin_bit_cast(unsigned short, b0) |
         ((unsigned int)__builtin_bit_cast(unsigned short, b1) << 16);
}

// K offset within a 32-wide k-block for fragment slot (half, j2) per the ISA
// 16-bit A/B layout: lanes 0-15 hold K=0..7 & 16..23, lanes 16-31 hold K=8..15 & 24..31.
__device__ __forceinline__ int frag_koff(int half, int j2) {
  return (j2 < 8) ? (8 * half + j2) : (16 + 8 * half + (j2 - 8));
}

// One-time: re-layout W [D,K] fp32 into WMMA B-fragment order, bf16, in global ws.
__global__ __launch_bounds__(256)
void wprep_kernel(const float* __restrict__ W, unsigned int* __restrict__ out) {
  const int idx = blockIdx.x * blockDim.x + threadIdx.x;
  if (idx >= NPREP) return;
  const int jd = idx & 7;
  const int L  = (idx >> 3) & 31;
  const int c  = idx >> 8;          // 0 .. NTILE*KB-1
  const int kb = c & (KB_ - 1);
  const int nt = c / KB_;
  const int half = L >> 4, ln = L & 15;
  const int ncol = nt * 16 + ln;
  const int k = kb * 32 + frag_koff(half, jd * 2);
  float f0 = 0.f, f1 = 0.f;
  if (ncol < K_) { f0 = W[k * K_ + ncol]; f1 = W[(k + 1) * K_ + ncol]; }
  out[idx] = pack_bf16(f0, f1);
}

// proj[b][k] = (mean_s tfh[b][s][:]) @ W[:,k]
__global__ __launch_bounds__(128)
void mean_proj_kernel(const float* __restrict__ tfh,
                      const float* __restrict__ W,    // [D,K]
                      float* __restrict__ proj) {     // [B,K]
  __shared__ float tmean[D_];
  const int b = blockIdx.x;
  const float4* tb4 = (const float4*)(tfh + (size_t)b * S_ * D_);
  for (int d4 = threadIdx.x; d4 < D_ / 4; d4 += blockDim.x) {
    float4 s = make_float4(0.f, 0.f, 0.f, 0.f);
    for (int j = 0; j < S_; ++j) {
      float4 v = tb4[j * (D_ / 4) + d4];
      s.x += v.x; s.y += v.y; s.z += v.z; s.w += v.w;
    }
    const float inv = 1.0f / (float)S_;
    ((float4*)tmean)[d4] = make_float4(s.x * inv, s.y * inv, s.z * inv, s.w * inv);
  }
  __syncthreads();
  for (int k = threadIdx.x; k < K_; k += blockDim.x) {
    float acc = 0.f;
    for (int d = 0; d < D_; ++d) acc += tmean[d] * W[d * K_ + k];
    proj[(size_t)b * K_ + k] = acc;
  }
}

// proj[b][k] = vec[b][:] @ W[:,k]
__global__ __launch_bounds__(128)
void vec_proj_kernel(const float* __restrict__ vec,   // [B,D]
                     const float* __restrict__ W,     // [D,K]
                     float* __restrict__ proj) {      // [B,K]
  __shared__ float sv[D_];
  const int b = blockIdx.x;
  const float* vb = vec + (size_t)b * D_;
  for (int d = threadIdx.x; d < D_; d += blockDim.x) sv[d] = vb[d];
  __syncthreads();
  for (int k = threadIdx.x; k < K_; k += blockDim.x) {
    float acc = 0.f;
    for (int d = 0; d < D_; ++d) acc += sv[d] * W[d * K_ + k];
    proj[(size_t)b * K_ + k] = acc;
  }
}

// Fused attention pass. One workgroup per batch, 7 waves (one N-tile each).
template<int ROWS, bool FINAL>
__global__ __launch_bounds__(ATTN_THREADS)
void attn_kernel(const float* __restrict__ feat,        // [B,ROWS,D]
                 const unsigned int* __restrict__ wPrep,// B-fragment-ordered bf16 W
                 const float* __restrict__ constProj,   // [B,K]
                 const float* __restrict__ wPi,         // [2K,1]
                 const float* __restrict__ bias,        // [ROWS]
                 const float* __restrict__ viBuf,       // [B,D] read iff FINAL
                 float* __restrict__ outBuf,            // [B,D]
                 int matOff, int constOff) {
  extern __shared__ float smem[];
  float*        sA   = smem;                         // MPAD*D fp32 staging (128 KB)
  unsigned int* sAbD = (unsigned int*)(sA + MPAD * D_); // bf16 A-fragments (64 KB)
  float* sScore = (float*)(sAbD + MPAD * D_ / 2);    // MPAD
  float* sP     = sScore + MPAD;                     // MPAD
  float* sConst = sP + MPAD;                         // 1

  const int tid = threadIdx.x;
  const int b   = blockIdx.x;
  const float* fb = feat + (size_t)b * ROWS * D_;

  if (tid < MPAD) sScore[tid] = 0.f;
  if (tid == 0)   *sConst = 0.f;

  float4* sA4 = (float4*)sA;
  const float4 z4 = make_float4(0.f, 0.f, 0.f, 0.f);

#if USE_TDM
  // --- Tensor Data Mover: DMA the ROWS*512 fp32 tile global -> LDS as a 1-D tensor.
  if (tid < 32) {  // one wave issues the per-wave DMA
    const unsigned long long ga = (unsigned long long)(uintptr_t)fb;
    const unsigned int ldsoff = (unsigned int)(uintptr_t)sA;      // LDS byte offset
    const unsigned int NE = (unsigned int)(ROWS * D_);            // elements (1-D tile)
    v4u g0;
    g0[0] = 1u;                                                   // count=1, user desc
    g0[1] = ldsoff;                                               // lds_addr
    g0[2] = (unsigned int)ga;                                     // global_addr[31:0]
    g0[3] = (unsigned int)((ga >> 32) & 0x01FFFFFFu) | (2u << 30);// addr[56:32] | type=2
    v8i g1;
    g1[0] = (int)(2u << 16);             // data_size=4B
    g1[1] = (int)((NE & 0xFFFFu) << 16); // tensor_dim0[15:0]
    g1[2] = (int)(NE >> 16);             // tensor_dim0[31:16]
    g1[3] = (int)(NE << 16);             // tile_dim0 = NE
    g1[4] = 0;                           // tile_dim1/2 = 0 (1-D)
    g1[5] = (int)NE;                     // stride0 (unused for 1-D)
    g1[6] = 0;  g1[7] = 0;
    v4i gz4 = {0, 0, 0, 0};
    v8i gz8 = {0, 0, 0, 0, 0, 0, 0, 0};
    __builtin_amdgcn_tensor_load_to_lds(g0, g1, gz4, gz4, gz8, 0);
  }
  // Zero the pad rows (disjoint LDS region) while the DMA is in flight.
  for (int i = ROWS * (D_ / 4) + tid; i < MPAD * (D_ / 4); i += blockDim.x) sA4[i] = z4;
  if (tid < 32) __builtin_amdgcn_s_wait_tensorcnt(0);
  __syncthreads();
#else
  {
    const int nvec = MPAD * (D_ / 4);
    const float4* fb4 = (const float4*)fb;
    for (int i = tid; i < nvec; i += blockDim.x) {
      const int row = i / (D_ / 4);
      sA4[i] = (row < ROWS) ? fb4[(size_t)row * (D_ / 4) + (i % (D_ / 4))] : z4;
    }
  }
  __syncthreads();
#endif

  // One-pass fp32 -> bf16 A-fragment re-layout (dword idx == output position).
  for (int idx = tid; idx < MPAD * D_ / 2; idx += blockDim.x) {
    const int jd = idx & 7;
    const int L  = (idx >> 3) & 31;
    const int c  = idx >> 8;           // (mt*KB + kb)
    const int kb = c & (KB_ - 1);
    const int mt = c >> 4;
    const int half = L >> 4, ln = L & 15;
    const int row = mt * 16 + ln;
    const int k = kb * 32 + frag_koff(half, jd * 2);
    sAbD[idx] = pack_bf16(sA[row * D_ + k], sA[row * D_ + k + 1]);
  }

  // Row-independent score term: sum_k tanh(constProj[b][k]) * wPi[constOff+k]
  if (tid < K_) {
    float h = tanhf(constProj[(size_t)b * K_ + tid]);
    atomicAdd(sConst, h * wPi[constOff + tid]);
  }
  __syncthreads();

  // --- WMMA GEMM: software-pipelined, fragments are straight 32-byte copies.
  const int lane = tid & 31;
  const int wave = tid >> 5;     // 0..6 == N-tile
  const int half = lane >> 4;
  const int ln   = lane & 15;
  const int nt   = wave;
  const int ncol = nt * 16 + ln;

  const v8u* aV = (const v8u*)sAbD;                  // [mt*KB+kb][lane] 32B elems
  const v8u* bV = (const v8u*)wPrep;                 // [nt*KB+kb][lane] 32B elems

  auto loadA = [&](int kb, int mt) {
    return __builtin_bit_cast(v16bf, aV[(mt * KB_ + kb) * 32 + lane]);
  };
  auto loadB = [&](int kb) {
    return __builtin_bit_cast(v16bf, bV[(nt * KB_ + kb) * 32 + lane]);
  };

  v8f acc0 = v8f{}, acc1 = v8f{}, acc2 = v8f{}, acc3 = v8f{};

  // Prologue: fetch k-block 0.
  v16bf bc = loadB(0);
  v16bf a0 = loadA(0, 0), a1 = loadA(0, 1), a2 = loadA(0, 2), a3 = loadA(0, 3);

#pragma unroll 2
  for (int kb = 0; kb < KB_; ++kb) {
    // Prefetch next k-block while the current WMMAs drain.
    const int nx = (kb + 1 < KB_) ? (kb + 1) : kb;
    v16bf bn = loadB(nx);
    v16bf n0 = loadA(nx, 0), n1 = loadA(nx, 1), n2 = loadA(nx, 2), n3 = loadA(nx, 3);
    acc0 = __builtin_amdgcn_wmma_f32_16x16x32_bf16(false, a0, false, bc, (short)0, acc0, false, false);
    acc1 = __builtin_amdgcn_wmma_f32_16x16x32_bf16(false, a1, false, bc, (short)0, acc1, false, false);
    acc2 = __builtin_amdgcn_wmma_f32_16x16x32_bf16(false, a2, false, bc, (short)0, acc2, false, false);
    acc3 = __builtin_amdgcn_wmma_f32_16x16x32_bf16(false, a3, false, bc, (short)0, acc3, false, false);
    bc = bn; a0 = n0; a1 = n1; a2 = n2; a3 = n3;
    SCHED_HINT();   // 2 VMEM-read, 8 DS-read, 4 WMMA per k-step
  }

  // Fused tanh + projection-row dot, reduced across N via LDS atomics.
  const float wp = (ncol < K_) ? wPi[matOff + ncol] : 0.f;
#pragma unroll
  for (int v = 0; v < 8; ++v) {
    const int mb = 8 * half + v;
    atomicAdd(&sScore[ 0 + mb], tanhf(acc0[v]) * wp);
    atomicAdd(&sScore[16 + mb], tanhf(acc1[v]) * wp);
    atomicAdd(&sScore[32 + mb], tanhf(acc2[v]) * wp);
    atomicAdd(&sScore[48 + mb], tanhf(acc3[v]) * wp);
  }
  __syncthreads();

  // Softmax over ROWS (tiny: serial on one lane)
  if (tid == 0) {
    const float c = *sConst;
    float mx = -1e30f;
    for (int m = 0; m < ROWS; ++m) {
      const float s = sScore[m] + c + bias[m];
      sScore[m] = s;
      mx = fmaxf(mx, s);
    }
    float sum = 0.f;
    for (int m = 0; m < ROWS; ++m) {
      const float e = expf(sScore[m] - mx);
      sP[m] = e;
      sum += e;
    }
    const float inv = 1.f / sum;
    for (int m = 0; m < ROWS; ++m) sP[m] *= inv;
  }
  __syncthreads();

  // Attention-weighted sum over rows (fp32 staging tile, exact)
  for (int d = tid; d < D_; d += blockDim.x) {
    float a = 0.f;
    for (int m = 0; m < ROWS; ++m) a += sP[m] * sA[m * D_ + d];
    if constexpr (FINAL)
      outBuf[(size_t)b * D_ + d] = viBuf[(size_t)b * D_ + d] + a;
    else
      outBuf[(size_t)b * D_ + d] = a;
  }
}

extern "C" void kernel_launch(void* const* d_in, const int* in_sizes, int n_in,
                              void* d_out, int out_size, void* d_ws, size_t ws_size,
                              hipStream_t stream) {
  const float* ifeat = (const float*)d_in[0];   // [B,R,D]
  const float* tfh   = (const float*)d_in[1];   // [B,S,D]
  const float* wVi0  = (const float*)d_in[2];   // [D,K]
  const float* wVt0  = (const float*)d_in[3];   // [D,K]
  const float* wPi0  = (const float*)d_in[4];   // [2K,1]
  const float* bPi0  = (const float*)d_in[5];   // [R]
  const float* wVi1  = (const float*)d_in[6];   // [D,K]
  const float* wVt1  = (const float*)d_in[7];   // [D,K]
  const float* wPi1  = (const float*)d_in[8];   // [2K,1]
  const float* bPi1  = (const float*)d_in[9];   // [S]
  float* out = (float*)d_out;

  float* ws    = (float*)d_ws;
  float* projT = ws;                          // [B,K]
  float* projI = projT + (size_t)B_ * K_;     // [B,K]
  float* viBuf = projI + (size_t)B_ * K_;     // [B,D]
  unsigned int* wprep0 = (unsigned int*)(viBuf + (size_t)B_ * D_); // NPREP dwords
  unsigned int* wprep1 = wprep0 + NPREP;                            // NPREP dwords

  const size_t shbytes = (size_t)(MPAD * D_ * 4 + MPAD * D_ * 2 + (2 * MPAD + 8) * 4);

  // One-time weight re-layout into WMMA B-fragment order (bf16).
  wprep_kernel<<<(NPREP + 255) / 256, 256, 0, stream>>>(wVi0, wprep0);
  wprep_kernel<<<(NPREP + 255) / 256, 256, 0, stream>>>(wVt1, wprep1);

  // Pass 0: mean text -> proj, fused image attention -> Vi
  mean_proj_kernel<<<B_, 128, 0, stream>>>(tfh, wVt0, projT);
  attn_kernel<R_, false><<<B_, ATTN_THREADS, shbytes, stream>>>(
      ifeat, wprep0, projT, wPi0, bPi0, nullptr, viBuf, /*matOff=*/0, /*constOff=*/K_);
  // Pass 1: Vi -> proj, fused text attention -> out = Vi + Vt
  vec_proj_kernel<<<B_, 128, 0, stream>>>(viBuf, wVi1, projI);
  attn_kernel<S_, true><<<B_, ATTN_THREADS, shbytes, stream>>>(
      tfh, wprep1, projI, wPi1, bPi1, viBuf, out, /*matOff=*/K_, /*constOff=*/0);
}
